// SelfAttention_25512105738886
// MI455X (gfx1250) — matrix operand order repaired and checked by reading, not compiled, not run on hardware
//
#include <hip/hip_runtime.h>
#include <hip/hip_bf16.h>

// Problem constants (match reference)
#define NB   2
#define SEQ  2048
#define EMB  1024
#define NH   16
#define HD   64   // EMB / NH

typedef _Float16 h16;
typedef __attribute__((ext_vector_type(16))) _Float16 v16h;
typedef __attribute__((ext_vector_type(8)))  _Float16 v8h;
typedef __attribute__((ext_vector_type(8)))  float    v8f;
typedef __attribute__((ext_vector_type(4)))  unsigned int u32x4;
typedef __attribute__((ext_vector_type(8)))  int      i32x8;
typedef __attribute__((ext_vector_type(4)))  int      i32x4;

#if __has_include(<hip/amd_detail/amd_gfx1250_TDM.h>)
#define TDM_6ARG 1
#else
#define TDM_6ARG 0
#endif

static __device__ __forceinline__ int rfl(int v) {
    return __builtin_amdgcn_readfirstlane(v);
}

// ---------------------------------------------------------------------------
// Fragment loaders per CDNA5 wave32 WMMA VGPR layouts (CDNA5 ISA)
//
// A (16x32 f16, MxK): lanes 0-15 -> M=lane, halfs[0..7]=K 0..7, halfs[8..15]=K 16..23
//                     lanes 16-31 -> M=lane-16, halfs[0..7]=K 8..15, halfs[8..15]=K 24..31
static __device__ __forceinline__
v16h load_frag_A(const h16* __restrict__ src, int ld, int row0, int k0, int lane) {
    const int hi = (lane >> 4) & 1;
    const int m  = row0 + (lane & 15);
    const h16* p = src + (size_t)m * ld + k0 + hi * 8;
    v8h lo = *(const v8h*)(p);
    v8h hh = *(const v8h*)(p + 16);
    return __builtin_shufflevector(lo, hh, 0,1,2,3,4,5,6,7,8,9,10,11,12,13,14,15);
}

// B (32x16 f16, KxN): lanes 0-15 hold K=0..7,16..23, lanes 16-31 hold K=8..15,24..31,
// column N = lane&15.  Source R is row-major with R[n][k] == B[k][n].
static __device__ __forceinline__
v16h load_frag_B(const h16* __restrict__ R, int ld, int col0, int k0, int lane) {
    const int hi = (lane >> 4) & 1;
    const int n  = col0 + (lane & 15);
    const h16* p = R + (size_t)n * ld + k0 + hi * 8;
    v8h lo = *(const v8h*)(p);
    v8h hh = *(const v8h*)(p + 16);
    return __builtin_shufflevector(lo, hh, 0,1,2,3,4,5,6,7,8,9,10,11,12,13,14,15);
}

static __device__ __forceinline__
v8f wmma_f16(v16h a, v16h b, v8f c) {
    return __builtin_amdgcn_wmma_f32_16x16x32_f16(false, a, false, b, (short)0, c,
                                                  false, false);
}

// ---------------------------------------------------------------------------
// Tensor Data Mover: 2D f16 tile load (tile_d1 rows x tile_d0 elems) global->LDS.
// D# packing per CDNA5 ISA.  All inputs wave-uniform.
static __device__ __forceinline__
void tdm_load_2d_f16(unsigned lds_addr, unsigned long long gaddr,
                     unsigned tensor_d0, unsigned tensor_d1,
                     unsigned tile_d0, unsigned tile_d1, unsigned stride0) {
    const unsigned glo = (unsigned)rfl((int)(unsigned)gaddr);
    const unsigned ghi = (unsigned)rfl((int)(unsigned)(gaddr >> 32));
    const unsigned lad = (unsigned)rfl((int)lds_addr);

    u32x4 g0;
    g0.x = 1u;                                        // count=1, user mode
    g0.y = lad;                                       // lds_addr (bytes)
    g0.z = glo;                                       // global_addr[31:0]
    g0.w = (ghi & 0x01FFFFFFu) | (2u << 30);          // global_addr[56:32] | type=2

    i32x8 g1;
    g1[0] = (int)(1u << 16);                          // workgroup_mask=0, data_size=1 (2B)
    g1[1] = (int)((tensor_d0 & 0xFFFFu) << 16);       // tensor_dim0[15:0]
    g1[2] = (int)(((tensor_d0 >> 16) & 0xFFFFu) |
                  ((tensor_d1 & 0xFFFFu) << 16));     // tensor_dim0[31:16] | tensor_dim1[15:0]
    g1[3] = (int)(((tensor_d1 >> 16) & 0xFFFFu) |
                  ((tile_d0 & 0xFFFFu) << 16));       // tensor_dim1[31:16] | tile_dim0
    g1[4] = (int)(tile_d1 & 0xFFFFu);                 // tile_dim1, tile_dim2=0
    g1[5] = (int)stride0;                             // tensor_dim0_stride[31:0]
    g1[6] = 0;                                        // stride0[47:32] | stride1[15:0]
    g1[7] = 0;                                        // stride1[47:16]

    i32x4 z4 = {0, 0, 0, 0};
#if TDM_6ARG
    i32x8 z8 = {0, 0, 0, 0, 0, 0, 0, 0};
    __builtin_amdgcn_tensor_load_to_lds(g0, g1, z4, z4, z8, 0);
#else
    __builtin_amdgcn_tensor_load_to_lds(g0, g1, z4, z4, 0);
#endif
}

// ---------------------------------------------------------------------------
__global__ void f32_to_f16_kernel(const float* __restrict__ in,
                                  h16* __restrict__ out, int n) {
    int i = blockIdx.x * blockDim.x + threadIdx.x;
    if (i < n) out[i] = (h16)in[i];
}

// ---------------------------------------------------------------------------
// 32x32 register-blocked GEMM: C[m,n] = sum_k X[m,k] * W[n,k]  (y = x @ W.T).
// 2 A-frags + 2 B-frags -> 4 WMMAs per 32-deep k-step (2x the AI of 16x16 tiles).
// "#pragma unroll 2" keeps the 4-WMMA body intact: full unroll made the
// allocator spill fragments to scratch (seen in round-2 asm).
// mode 0: out[((nb*NH+h)*SEQ + s)*HD + d]   (head-major rows,   for Q, K)
// mode 1: out[((nb*NH+h)*HD + d)*SEQ + s]   (per-head transpose, for V)
static __device__ __forceinline__
void store_proj_tile(h16* __restrict__ out, int mode, int row0, int col0,
                     v8f acc, int lane) {
    const int hi = (lane >> 4) & 1;
    const int cn = col0 + (lane & 15);
    const int h  = cn / HD, d = cn % HD;
    for (int j = 0; j < 8; ++j) {
        int m  = row0 + j + 8 * hi;
        int nb = m / SEQ, s = m % SEQ;
        size_t idx = (mode == 0)
            ? ((size_t)((nb * NH + h) * SEQ + s)) * HD + d
            : ((size_t)((nb * NH + h) * HD  + d)) * SEQ + s;
        out[idx] = (h16)acc[j];
    }
}

__global__ void __launch_bounds__(128, 1)
proj_gemm_kernel(const h16* __restrict__ X,
                 const h16* __restrict__ W,
                 h16* __restrict__ out, int mode) {
    const int lane = threadIdx.x & 31;
    const int wave = threadIdx.x >> 5;
    const int tile = blockIdx.x * (blockDim.x >> 5) + wave;
    const int ntn  = EMB / 32;                 // 32 column supertiles
    const int r0   = (tile / ntn) * 32;
    const int c0   = (tile % ntn) * 32;

    v8f a00 = {}, a01 = {}, a10 = {}, a11 = {};
#pragma unroll 2
    for (int k0 = 0; k0 < EMB; k0 += 32) {
        v16h A0 = load_frag_A(X, EMB, r0,      k0, lane);
        v16h A1 = load_frag_A(X, EMB, r0 + 16, k0, lane);
        v16h B0 = load_frag_B(W, EMB, c0,      k0, lane);
        v16h B1 = load_frag_B(W, EMB, c0 + 16, k0, lane);
        a00 = wmma_f16(A0, B0, a00);
        a01 = wmma_f16(A0, B1, a01);
        a10 = wmma_f16(A1, B0, a10);
        a11 = wmma_f16(A1, B1, a11);
    }
    store_proj_tile(out, mode, r0,      c0,      a00, lane);
    store_proj_tile(out, mode, r0,      c0 + 16, a01, lane);
    store_proj_tile(out, mode, r0 + 16, c0,      a10, lane);
    store_proj_tile(out, mode, r0 + 16, c0 + 16, a11, lane);
}

// ---------------------------------------------------------------------------
// Final projection with bias, f32 output (same 32x32 blocking).
__global__ void __launch_bounds__(128, 1)
out_gemm_kernel(const h16* __restrict__ X,
                const h16* __restrict__ W,
                const float* __restrict__ bias,
                float* __restrict__ out) {
    const int lane = threadIdx.x & 31;
    const int wave = threadIdx.x >> 5;
    const int tile = blockIdx.x * (blockDim.x >> 5) + wave;
    const int ntn  = EMB / 32;
    const int r0   = (tile / ntn) * 32;
    const int c0   = (tile % ntn) * 32;

    v8f a00 = {}, a01 = {}, a10 = {}, a11 = {};
#pragma unroll 2
    for (int k0 = 0; k0 < EMB; k0 += 32) {
        v16h A0 = load_frag_A(X, EMB, r0,      k0, lane);
        v16h A1 = load_frag_A(X, EMB, r0 + 16, k0, lane);
        v16h B0 = load_frag_B(W, EMB, c0,      k0, lane);
        v16h B1 = load_frag_B(W, EMB, c0 + 16, k0, lane);
        a00 = wmma_f16(A0, B0, a00);
        a01 = wmma_f16(A0, B1, a01);
        a10 = wmma_f16(A1, B0, a10);
        a11 = wmma_f16(A1, B1, a11);
    }

    const int hi = (lane >> 4) & 1;
    const int cl = lane & 15;
    const float b0 = bias[c0 + cl];
    const float b1 = bias[c0 + 16 + cl];
    for (int j = 0; j < 8; ++j) {
        int m0 = r0 + j + 8 * hi;
        out[(size_t)m0 * EMB + c0 + cl]             = a00[j] + b0;
        out[(size_t)m0 * EMB + c0 + 16 + cl]        = a01[j] + b1;
        out[(size_t)(m0 + 16) * EMB + c0 + cl]      = a10[j] + b0;
        out[(size_t)(m0 + 16) * EMB + c0 + 16 + cl] = a11[j] + b1;
    }
}

// ---------------------------------------------------------------------------
// Flash attention per (nb, h, 16-row q tile); one wave per tile, 4 waves/block.
// K tiles (32 x 64 f16 = 4KB) are double-buffered into LDS by the Tensor Data
// Mover, overlapping the DMA of tile i+1 with softmax+WMMA work on tile i.
__global__ void __launch_bounds__(128, 1)
attn_kernel(const h16* __restrict__ Qp,
            const h16* __restrict__ Kp,
            const h16* __restrict__ VT,
            const int* __restrict__ mask,
            h16* __restrict__ AO) {
    __shared__ __align__(32) h16 kbuf[4][2][32 * HD];   // per-wave K tiles (2x4KB)
    __shared__ __align__(32) h16 pbuf[4][16 * 32];      // per-wave P tile (1KB)

    const int lane = threadIdx.x & 31;
    const int wave = threadIdx.x >> 5;
    const int wid  = blockIdx.x * (blockDim.x >> 5) + wave;

    const int qtiles = SEQ / 16;
    const int qt = wid % qtiles;
    const int nh = wid / qtiles;
    const int h  = nh % NH, nb = nh / NH;
    const int q0 = qt * 16;

    const h16* Qh = Qp + (size_t)(nb * NH + h) * SEQ * HD;
    const h16* Kh = Kp + (size_t)(nb * NH + h) * SEQ * HD;
    const h16* Vh = VT + (size_t)(nb * NH + h) * HD  * SEQ;
    const int* Mh = mask + (size_t)(nb * NH + h) * SEQ * SEQ;

    // Q tile fragments live in registers across the whole row sweep (D=64 -> 2 frags)
    const v16h qa0 = load_frag_A(Qh, HD, q0, 0,  lane);
    const v16h qa1 = load_frag_A(Qh, HD, q0, 32, lane);

    const int hi = (lane >> 4) & 1;
    const int cl = lane & 15;

    float mrow[8], lrow[8];
    v8f o0 = {}, o1 = {}, o2 = {}, o3 = {};
    for (int j = 0; j < 8; ++j) { mrow[j] = -3.0e38f; lrow[j] = 0.0f; }

    const float scale = 0.022097086912079608f;   // 1/sqrt(2048): key-seq-length scaling

    const unsigned kslab = (unsigned)(uintptr_t)&kbuf[wave][0][0];
    const unsigned long long kbase = (unsigned long long)(uintptr_t)Kh;
    const unsigned kTileBytes = 32u * HD * 2u;   // 4KB per buffer

    // prologue: TDM-load tile 0 into buffer 0
    tdm_load_2d_f16(kslab, kbase, HD, SEQ, HD, 32, HD);

#pragma unroll 1
    for (int kb = 0; kb < SEQ; kb += 32) {
        const int cur = (kb >> 5) & 1;
        if (kb + 32 < SEQ) {
            // issue DMA for next tile, then wait only for the current one
            tdm_load_2d_f16(kslab + (cur ^ 1) * kTileBytes,
                            kbase + (unsigned long long)(kb + 32) * HD * 2u,
                            HD, SEQ, HD, 32, HD);
            __builtin_amdgcn_s_wait_tensorcnt(1);
            // stream mask lines for next iteration ahead of the softmax
            __builtin_prefetch(&Mh[(size_t)(q0 + cl) * SEQ + kb + 32 + hi * 16], 0, 0);
        } else {
            __builtin_amdgcn_s_wait_tensorcnt(0);
        }
        const h16* Kt = &kbuf[wave][cur][0];

        // --- energy: two 16x16 tiles over 32 key columns, K-loop over D=64
        v8f e0 = {}, e1 = {};
        e0 = wmma_f16(qa0, load_frag_B(Kt, HD, 0,  0,  lane), e0);
        e0 = wmma_f16(qa1, load_frag_B(Kt, HD, 0,  32, lane), e0);
        e1 = wmma_f16(qa0, load_frag_B(Kt, HD, 16, 0,  lane), e1);
        e1 = wmma_f16(qa1, load_frag_B(Kt, HD, 16, 32, lane), e1);

        // --- mask + scale (reference: where(mask==0, -1e20, e) / sqrt(S))
        float s0[8], s1[8], vmax[8], vsum[8];
        const int c0 = kb + cl, c1 = kb + 16 + cl;
        for (int j = 0; j < 8; ++j) {
            int row = q0 + j + 8 * hi;
            float x0 = (Mh[(size_t)row * SEQ + c0] == 0) ? -1.0e20f : e0[j];
            float x1 = (Mh[(size_t)row * SEQ + c1] == 0) ? -1.0e20f : e1[j];
            s0[j] = x0 * scale;
            s1[j] = x1 * scale;
            vmax[j] = fmaxf(s0[j], s1[j]);
        }
        // row-max across the 16 lanes of each half-wave (rows j / j+8 split)
        for (int off = 1; off < 16; off <<= 1)
            for (int j = 0; j < 8; ++j)
                vmax[j] = fmaxf(vmax[j], __shfl_xor(vmax[j], off, 32));

        float corr[8];
        for (int j = 0; j < 8; ++j) {
            float mn = fmaxf(mrow[j], vmax[j]);
            corr[j]  = __expf(mrow[j] - mn);
            mrow[j]  = mn;
            s0[j]    = __expf(s0[j] - mn);
            s1[j]    = __expf(s1[j] - mn);
            vsum[j]  = s0[j] + s1[j];
        }
        for (int off = 1; off < 16; off <<= 1)
            for (int j = 0; j < 8; ++j)
                vsum[j] += __shfl_xor(vsum[j], off, 32);
        for (int j = 0; j < 8; ++j)
            lrow[j] = lrow[j] * corr[j] + vsum[j];

        // rescale accumulators (same row mapping as C layout)
        for (int j = 0; j < 8; ++j) {
            o0[j] *= corr[j]; o1[j] *= corr[j];
            o2[j] *= corr[j]; o3[j] *= corr[j];
        }

        // --- C-layout -> A-fragment re-shape of P through per-wave LDS slab
        h16* P = pbuf[wave];
        for (int j = 0; j < 8; ++j) {
            int row = j + 8 * hi;
            P[row * 32 + cl]      = (h16)s0[j];
            P[row * 32 + 16 + cl] = (h16)s1[j];
        }
        __builtin_amdgcn_wave_barrier();   // keep ds stores before ds loads
        v16h pa = load_frag_A(P, 32, 0, 0, lane);

        // --- O += P * V  (V^T rows are contiguous over s)
        o0 = wmma_f16(pa, load_frag_B(Vh, SEQ,  0, kb, lane), o0);
        o1 = wmma_f16(pa, load_frag_B(Vh, SEQ, 16, kb, lane), o1);
        o2 = wmma_f16(pa, load_frag_B(Vh, SEQ, 32, kb, lane), o2);
        o3 = wmma_f16(pa, load_frag_B(Vh, SEQ, 48, kb, lane), o3);
    }

    // epilogue: normalize and store to AO[nb][s][h][d] (flat [m][e] for out-proj)
    for (int j = 0; j < 8; ++j) {
        int row = q0 + j + 8 * hi;
        float inv = 1.0f / lrow[j];
        size_t base = ((size_t)(nb * SEQ + row) * NH + h) * HD + cl;
        AO[base +  0] = (h16)(o0[j] * inv);
        AO[base + 16] = (h16)(o1[j] * inv);
        AO[base + 32] = (h16)(o2[j] * inv);
        AO[base + 48] = (h16)(o3[j] * inv);
    }
}

// ---------------------------------------------------------------------------
extern "C" void kernel_launch(void* const* d_in, const int* in_sizes, int n_in,
                              void* d_out, int out_size, void* d_ws, size_t ws_size,
                              hipStream_t stream) {
    const float* q    = (const float*)d_in[0];
    const float* k    = (const float*)d_in[1];
    const float* v    = (const float*)d_in[2];
    const int*   mask = (const int*)  d_in[3];
    const float* Wq   = (const float*)d_in[4];
    const float* Wk   = (const float*)d_in[5];
    const float* Wv   = (const float*)d_in[6];
    const float* Wo   = (const float*)d_in[7];
    const float* bo   = (const float*)d_in[8];

    const size_t NSE = (size_t)NB * SEQ * EMB;   // 4,194,304
    const size_t EE  = (size_t)EMB * EMB;        // 1,048,576

    // workspace carve-up (halfs): 7*NSE + 4*EE = 32Mi halfs = 64 MB
    h16* ws   = (h16*)d_ws;
    h16* Xq   = ws;
    h16* Xk   = Xq  + NSE;
    h16* Xv   = Xk  + NSE;
    h16* W16q = Xv  + NSE;
    h16* W16k = W16q + EE;
    h16* W16v = W16k + EE;
    h16* W16o = W16v + EE;
    h16* Qp   = W16o + EE;   // [nb][h][s][d]
    h16* Kp   = Qp  + NSE;   // [nb][h][s][d]
    h16* VTp  = Kp  + NSE;   // [nb][h][d][s]
    h16* AO   = VTp + NSE;   // [nb][s][h][d] == flat [m][e]

    // 1) convert inputs + weights to f16
    const int cb = 256;
    f32_to_f16_kernel<<<((int)NSE + cb - 1) / cb, cb, 0, stream>>>(q,  Xq,   (int)NSE);
    f32_to_f16_kernel<<<((int)NSE + cb - 1) / cb, cb, 0, stream>>>(k,  Xk,   (int)NSE);
    f32_to_f16_kernel<<<((int)NSE + cb - 1) / cb, cb, 0, stream>>>(v,  Xv,   (int)NSE);
    f32_to_f16_kernel<<<((int)EE  + cb - 1) / cb, cb, 0, stream>>>(Wq, W16q, (int)EE);
    f32_to_f16_kernel<<<((int)EE  + cb - 1) / cb, cb, 0, stream>>>(Wk, W16k, (int)EE);
    f32_to_f16_kernel<<<((int)EE  + cb - 1) / cb, cb, 0, stream>>>(Wv, W16v, (int)EE);
    f32_to_f16_kernel<<<((int)EE  + cb - 1) / cb, cb, 0, stream>>>(Wo, W16o, (int)EE);

    // 2) Q/K/V projections: 4096 32x32 supertiles, 4 waves (128 thr) per block
    const int gemm_tiles = (NB * SEQ / 32) * (EMB / 32);   // 4096
    dim3 blk(128);
    proj_gemm_kernel<<<gemm_tiles / 4, blk, 0, stream>>>(Xq, W16q, Qp,  0);
    proj_gemm_kernel<<<gemm_tiles / 4, blk, 0, stream>>>(Xk, W16k, Kp,  0);
    proj_gemm_kernel<<<gemm_tiles / 4, blk, 0, stream>>>(Xv, W16v, VTp, 1);

    // 3) flash attention: one wave per (nb, h, q-tile) = 4096 waves
    const int attn_waves = NB * NH * (SEQ / 16);
    attn_kernel<<<attn_waves / 4, blk, 0, stream>>>(Qp, Kp, VTp, mask, AO);

    // 4) output projection + bias -> f32 d_out
    out_gemm_kernel<<<gemm_tiles / 4, blk, 0, stream>>>(AO, W16o, bo, (float*)d_out);
}
